// Qwen3VLVisionModel_69922067578963
// MI455X (gfx1250) — matrix operand order repaired and checked by reading, not compiled, not run on hardware
//
#include <hip/hip_runtime.h>

typedef __attribute__((ext_vector_type(16))) __bf16 v16bf;
typedef __attribute__((ext_vector_type(8)))  float  v8f;
typedef __attribute__((ext_vector_type(4)))  unsigned int u32x4;
typedef __attribute__((ext_vector_type(4)))  int    i32x4;
typedef __attribute__((ext_vector_type(8)))  int    i32x8;

#define T_TOK 2048
#define DMODEL 1152
#define NHEAD 16
#define HDIM 72
#define HPAD 96
#define VPAD 80
#define FFDIM 4608
#define DEPTH 4
#define MERGED_D 4608
#define OUT_D 3584
#define KC 128           // K-chunk staged per TDM transfer
#define BN 64            // N rows per block tile

// ---------------------------------------------------------------- helpers
static __device__ inline v8f vzero() {
    v8f z = {0.f,0.f,0.f,0.f,0.f,0.f,0.f,0.f};
    return z;
}

static __device__ inline v8f wmma_bf16(v16bf a, v16bf b, v8f c) {
    // D = A(16x32 bf16) * B(32x16 bf16) + C(16x16 f32)
    return __builtin_amdgcn_wmma_f32_16x16x32_bf16(
        false, a, false, b, (short)0, c, false, false);
}

// A-matrix 16x32 bf16 fragment (ISA 7.12.2): lane m = lane&15,
// K elements {8h..8h+7} and {16+8h..16+8h+7}, h = lane>>4.
static __device__ inline v16bf load_fragA(const __bf16* base, int row_stride,
                                          int lane, int kbase) {
    const __bf16* p = base + (size_t)(lane & 15) * row_stride + kbase + ((lane >> 4) * 8);
    union { v16bf v; uint4 u[2]; } r;
    r.u[0] = *(const uint4*)(p);
    r.u[1] = *(const uint4*)(p + 16);
    return r.v;
}

// B-matrix 32x16 bf16 fragment: lane n = lane&15, K {8h..8h+7} and {16+8h..16+8h+7}.
// base points to Bt (N x K, row-major) row block; row = n.
static __device__ inline v16bf load_fragB(const __bf16* base, int row_stride,
                                          int lane, int kbase) {
    const __bf16* p = base + (size_t)(lane & 15) * row_stride + kbase + ((lane >> 4) * 8);
    union { v16bf v; uint4 u[2]; } r;
    r.u[0] = *(const uint4*)(p);
    r.u[1] = *(const uint4*)(p + 16);
    return r.v;
}

// Tensor Data Mover: DMA a 2-D tile (rows x kc bf16 elements, row stride K)
// from global (gsrc = tile start) into LDS at byte offset lds_off.
// D# layout per CDNA5 ISA sec. 8.3-8.6.
static __device__ inline void tdm_load_tile(const __bf16* gsrc, unsigned int lds_off,
                                            int K, int N, int rows, int kc) {
    unsigned long long ga = (unsigned long long)(size_t)gsrc;
    u32x4 g0;
    g0.x = 1u;                                   // count=1, user descriptor
    g0.y = lds_off;                              // lds_addr (bytes)
    g0.z = (unsigned int)ga;                     // global_addr[31:0]
    g0.w = (unsigned int)((ga >> 32) & 0x01FFFFFFull) | (2u << 30);  // addr[56:32] | type=2
    i32x8 g1;
    g1[0] = 1 << 16;                             // workgroup_mask=0, data_size=1 (2 bytes)
    g1[1] = (K & 0xFFFF) << 16;                  // tensor_dim0[15:0] (atomic addr = 0)
    g1[2] = ((K >> 16) & 0xFFFF) | ((N & 0xFFFF) << 16);   // tensor_dim0[31:16] | tensor_dim1[15:0]
    g1[3] = ((N >> 16) & 0xFFFF) | (kc << 16);   // tensor_dim1[31:16] | tile_dim0
    g1[4] = rows;                                // tile_dim1 | tile_dim2=0
    g1[5] = K;                                   // tensor_dim0_stride[31:0] (elements)
    g1[6] = 0;                                   // stride hi | dim1_stride lo (unused, 2-D)
    g1[7] = 0;
    i32x4 g2 = {0,0,0,0};
    i32x4 g3 = {0,0,0,0};
#if defined(__clang_major__) && __clang_major__ >= 23
    i32x8 g4 = {0,0,0,0,0,0,0,0};
    __builtin_amdgcn_tensor_load_to_lds(g0, g1, g2, g3, g4, 0);
#else
    __builtin_amdgcn_tensor_load_to_lds(g0, g1, g2, g3, 0);
#endif
}

static __device__ inline float gelu_tanh(float x) {
    float x3 = x * x * x;
    return 0.5f * x * (1.f + tanhf(0.7978845608028654f * (x + 0.044715f * x3)));
}
static __device__ inline float gelu_erf(float x) {
    return 0.5f * x * (1.f + erff(x * 0.7071067811865475f));
}

// ------------------------------------------------------- weight transpose
// src: f32 [K][N] -> dst: bf16 [N][K]
__global__ void transpose_bf16_kernel(const float* __restrict__ src,
                                      __bf16* __restrict__ dst, int K, int N) {
    __shared__ float tile[32][33];
    int kb = blockIdx.x * 32, nb = blockIdx.y * 32;
    int tx = threadIdx.x, ty = threadIdx.y;
#pragma unroll
    for (int i = 0; i < 32; i += 8)
        tile[ty + i][tx] = src[(size_t)(kb + ty + i) * N + nb + tx];
    __syncthreads();
#pragma unroll
    for (int i = 0; i < 32; i += 8)
        dst[(size_t)(nb + ty + i) * K + kb + tx] = (__bf16)tile[tx][ty + i];
}

// ------------------------------------------------------------ layernorm
__global__ __launch_bounds__(256)
void ln_kernel(const float* __restrict__ h, const float* __restrict__ w,
               const float* __restrict__ b, __bf16* __restrict__ z, int Dn) {
    __shared__ float red[256];
    int row = blockIdx.x, tid = threadIdx.x;
    const float* hr = h + (size_t)row * Dn;
    float s = 0.f;
    for (int i = tid; i < Dn; i += 256) s += hr[i];
    red[tid] = s; __syncthreads();
    for (int st = 128; st > 0; st >>= 1) {
        if (tid < st) red[tid] += red[tid + st];
        __syncthreads();
    }
    float mean = red[0] / (float)Dn;
    __syncthreads();
    float v = 0.f;
    for (int i = tid; i < Dn; i += 256) { float d = hr[i] - mean; v += d * d; }
    red[tid] = v; __syncthreads();
    for (int st = 128; st > 0; st >>= 1) {
        if (tid < st) red[tid] += red[tid + st];
        __syncthreads();
    }
    float rstd = rsqrtf(red[0] / (float)Dn + 1e-6f);
    for (int i = tid; i < Dn; i += 256)
        z[(size_t)row * Dn + i] = (__bf16)((hr[i] - mean) * rstd * w[i] + b[i]);
}

// ------------------------------------------------------------- WMMA GEMM
// C[M][N] = act(A_bf16[MxK] @ Bt_bf16[NxK]^T + bias) (+ residual)
// 4 waves / block, 16x64 tile per wave. The 64xKC bf16 B-panel is DMA'd
// into LDS by the Tensor Data Mover once per block (not once per wave) and
// double-buffered against TENSORcnt so the next panel transfers while the
// current one feeds 16 WMMAs.
__global__ __launch_bounds__(128)
void gemm_kernel(const __bf16* __restrict__ A, const __bf16* __restrict__ Bt,
                 const float* __restrict__ bias, const float* __restrict__ res,
                 float* __restrict__ Cf, __bf16* __restrict__ Cb,
                 int M, int N, int K, int act) {
    __shared__ __align__(16) __bf16 bpan[2][BN * KC];  // 2 x 16KB
    int lane = threadIdx.x & 31;
    int wave = threadIdx.x >> 5;
    int m0 = blockIdx.y * 64 + wave * 16;
    int n0 = blockIdx.x * BN;
    v8f acc[4];
#pragma unroll
    for (int j = 0; j < 4; ++j) acc[j] = vzero();

    const __bf16* bt_blk = Bt + (size_t)n0 * K;
    if (wave == 0)
        tdm_load_tile(bt_blk, (unsigned int)(size_t)(&bpan[0][0]), K, N, BN, KC);

    for (int c0 = 0; c0 < K; c0 += KC) {
        int cur = (c0 / KC) & 1;
        if (wave == 0) {
            if (c0 + KC < K) {
                tdm_load_tile(bt_blk + (c0 + KC),
                              (unsigned int)(size_t)(&bpan[cur ^ 1][0]), K, N, BN, KC);
                __builtin_amdgcn_s_wait_tensorcnt((short)1);  // panel c0 landed
            } else {
                __builtin_amdgcn_s_wait_tensorcnt((short)0);
            }
        }
        __syncthreads();  // panel visible to all waves
        const __bf16* bp = &bpan[cur][0];
#pragma unroll
        for (int kk = 0; kk < KC; kk += 32) {
            v16bf a = load_fragA(A + (size_t)m0 * K, K, lane, c0 + kk);
#pragma unroll
            for (int j = 0; j < 4; ++j) {
                v16bf b = load_fragB(bp + (size_t)(j * 16) * KC, KC, lane, kk);
                acc[j] = wmma_bf16(a, b, acc[j]);
            }
        }
        __syncthreads();  // all waves done with panel before it is overwritten
    }

    int half = lane >> 4;
#pragma unroll
    for (int j = 0; j < 4; ++j) {
        int nn = n0 + j * 16 + (lane & 15);
        float bv = bias ? bias[nn] : 0.f;
#pragma unroll
        for (int r = 0; r < 8; ++r) {
            int mm = m0 + r + 8 * half;
            float c = acc[j][r] + bv;
            if (act == 1) c = gelu_tanh(c);
            else if (act == 2) c = gelu_erf(c);
            if (res) c += res[(size_t)mm * N + nn];
            if (Cf) Cf[(size_t)mm * N + nn] = c;
            if (Cb) Cb[(size_t)mm * N + nn] = (__bf16)c;
        }
    }
}

// ------------------------------------------------------------------ RoPE
// qkv f32 [T][3456] -> qp,kp bf16 [NH][T][96] (zero pad), vt bf16 [NH][80][T]
__global__ void rope_kernel(const float* __restrict__ qkv,
                            const float* __restrict__ cosb,
                            const float* __restrict__ sinb,
                            __bf16* __restrict__ qp, __bf16* __restrict__ kp,
                            __bf16* __restrict__ vt) {
    int t = blockIdx.x, h = blockIdx.y, i = threadIdx.x;  // i in [0,96)
    float qv = 0.f, kv = 0.f;
    if (i < HDIM) {
        float c = cosb[(size_t)t * HDIM + i];
        float s = sinb[(size_t)t * HDIM + i];
        size_t base = (size_t)t * (3 * DMODEL) + h * HDIM;
        float q = qkv[base + i];
        float k = qkv[base + DMODEL + i];
        float qr = (i < 36) ? -qkv[base + i + 36] : qkv[base + i - 36];
        float kr = (i < 36) ? -qkv[base + DMODEL + i + 36] : qkv[base + DMODEL + i - 36];
        qv = q * c + qr * s;
        kv = k * c + kr * s;
    }
    size_t o = ((size_t)h * T_TOK + t) * HPAD + i;
    qp[o] = (__bf16)qv;
    kp[o] = (__bf16)kv;
    if (i < VPAD) {
        float v = (i < HDIM) ? qkv[(size_t)t * (3 * DMODEL) + 2 * DMODEL + h * HDIM + i] : 0.f;
        vt[((size_t)h * VPAD + i) * T_TOK + t] = (__bf16)v;
    }
}

// ------------------------------------------------------- flash attention
// one wave per (16-query tile, head); segments are 32-aligned so 32-key
// chunks are uniform-segment and skipping == -1e30 mask.
__global__ __launch_bounds__(32)
void attn_kernel(const __bf16* __restrict__ qp, const __bf16* __restrict__ kp,
                 const __bf16* __restrict__ vt, const int* __restrict__ seg,
                 __bf16* __restrict__ ao) {
    __shared__ __align__(16) __bf16 pb[16 * 32];
    int lane = threadIdx.x;
    int qt = blockIdx.x, h = blockIdx.y;
    int q0 = qt * 16;
    int n = lane & 15, half = lane >> 4;

    v16bf aq[3];
#pragma unroll
    for (int kk = 0; kk < 3; ++kk)
        aq[kk] = load_fragA(qp + ((size_t)h * T_TOK + q0) * HPAD, HPAD, lane, kk * 32);

    int sq = seg[q0];
    float rm[8], rs[8];
    v8f o[5];
#pragma unroll
    for (int r = 0; r < 8; ++r) { rm[r] = -1e30f; rs[r] = 0.f; }
#pragma unroll
    for (int j = 0; j < 5; ++j) o[j] = vzero();

    const float sm = 0.11785113019775793f;  // 72^-0.5

    for (int k0 = 0; k0 < T_TOK; k0 += 32) {
        if (seg[k0] != sq) continue;  // uniform branch, EXEC stays full
        v8f s0 = vzero(), s1 = vzero();
#pragma unroll
        for (int kk = 0; kk < 3; ++kk) {
            v16bf b0 = load_fragB(kp + ((size_t)h * T_TOK + k0) * HPAD, HPAD, lane, kk * 32);
            v16bf b1 = load_fragB(kp + ((size_t)h * T_TOK + k0 + 16) * HPAD, HPAD, lane, kk * 32);
            s0 = wmma_bf16(aq[kk], b0, s0);
            s1 = wmma_bf16(aq[kk], b1, s1);
        }
#pragma unroll
        for (int r = 0; r < 8; ++r) {
            float a0 = s0[r] * sm, a1 = s1[r] * sm;
            float t = fmaxf(a0, a1);
            t = fmaxf(t, __shfl_xor(t, 1, 32));
            t = fmaxf(t, __shfl_xor(t, 2, 32));
            t = fmaxf(t, __shfl_xor(t, 4, 32));
            t = fmaxf(t, __shfl_xor(t, 8, 32));
            float mn = fmaxf(rm[r], t);
            float cf = __expf(rm[r] - mn);
            float p0 = __expf(a0 - mn), p1 = __expf(a1 - mn);
            float ps = p0 + p1;
            ps += __shfl_xor(ps, 1, 32);
            ps += __shfl_xor(ps, 2, 32);
            ps += __shfl_xor(ps, 4, 32);
            ps += __shfl_xor(ps, 8, 32);
            rs[r] = rs[r] * cf + ps;
            rm[r] = mn;
#pragma unroll
            for (int j = 0; j < 5; ++j) o[j][r] *= cf;
            pb[(r + 8 * half) * 32 + n] = (__bf16)p0;
            pb[(r + 8 * half) * 32 + 16 + n] = (__bf16)p1;
        }
        __syncthreads();
        v16bf pa = load_fragA(pb, 32, lane, 0);
#pragma unroll
        for (int j = 0; j < 5; ++j) {
            v16bf bv = load_fragB(vt + ((size_t)h * VPAD + j * 16) * T_TOK, T_TOK, lane, k0);
            o[j] = wmma_bf16(pa, bv, o[j]);
        }
        __syncthreads();
    }
#pragma unroll
    for (int j = 0; j < 5; ++j) {
        int d = j * 16 + n;
        if (d < HDIM) {
#pragma unroll
            for (int r = 0; r < 8; ++r) {
                int m = q0 + r + 8 * half;
                ao[(size_t)m * DMODEL + h * HDIM + d] = (__bf16)(o[j][r] / rs[r]);
            }
        }
    }
}

// ------------------------------------------------------------------ host
extern "C" void kernel_launch(void* const* d_in, const int* in_sizes, int n_in,
                              void* d_out, int out_size, void* d_ws, size_t ws_size,
                              hipStream_t stream) {
    (void)in_sizes; (void)n_in; (void)out_size; (void)ws_size;
    const float* x        = (const float*)d_in[0];
    const float* cosb     = (const float*)d_in[1];
    const float* sinb     = (const float*)d_in[2];
    const int*   seg      = (const int*)  d_in[3];
    const float* ln1w     = (const float*)d_in[4];
    const float* ln1b     = (const float*)d_in[5];
    const float* wqkv     = (const float*)d_in[6];
    const float* bqkv     = (const float*)d_in[7];
    const float* wproj    = (const float*)d_in[8];
    const float* bproj    = (const float*)d_in[9];
    const float* ln2w     = (const float*)d_in[10];
    const float* ln2b     = (const float*)d_in[11];
    const float* wfc1     = (const float*)d_in[12];
    const float* bfc1     = (const float*)d_in[13];
    const float* wfc2     = (const float*)d_in[14];
    const float* bfc2     = (const float*)d_in[15];
    const float* mlnw     = (const float*)d_in[16];
    const float* mlnb     = (const float*)d_in[17];
    const float* mw1      = (const float*)d_in[18];
    const float* mb1      = (const float*)d_in[19];
    const float* mw2      = (const float*)d_in[20];
    const float* mb2      = (const float*)d_in[21];

    char* wsp = (char*)d_ws;
    size_t off = 0;
    auto alloc = [&](size_t bytes) {
        char* p = wsp + off;
        off += (bytes + 255) & ~(size_t)255;
        return p;
    };
    float*  h   = (float*) alloc((size_t)T_TOK * DMODEL * 4);
    __bf16* z   = (__bf16*)alloc((size_t)T_TOK * DMODEL * 2);
    float*  qkv = (float*) alloc((size_t)T_TOK * 3 * DMODEL * 4);
    __bf16* qp  = (__bf16*)alloc((size_t)NHEAD * T_TOK * HPAD * 2);
    __bf16* kp  = (__bf16*)alloc((size_t)NHEAD * T_TOK * HPAD * 2);
    __bf16* vt  = (__bf16*)alloc((size_t)NHEAD * VPAD * T_TOK * 2);
    __bf16* ao  = (__bf16*)alloc((size_t)T_TOK * DMODEL * 2);
    __bf16* g   = (__bf16*)alloc((size_t)T_TOK * FFDIM * 2);
    __bf16* wt  = (__bf16*)alloc((size_t)MERGED_D * MERGED_D * 2);  // reused

    hipMemcpyAsync(h, x, (size_t)T_TOK * DMODEL * sizeof(float),
                   hipMemcpyDeviceToDevice, stream);

    dim3 tb(32, 8);
    for (int d = 0; d < DEPTH; ++d) {
        // ln1 -> z
        ln_kernel<<<T_TOK, 256, 0, stream>>>(h, ln1w + d * DMODEL, ln1b + d * DMODEL, z, DMODEL);
        // qkv = z @ wqkv + bqkv
        transpose_bf16_kernel<<<dim3(DMODEL / 32, 3 * DMODEL / 32), tb, 0, stream>>>(
            wqkv + (size_t)d * DMODEL * 3 * DMODEL, wt, DMODEL, 3 * DMODEL);
        gemm_kernel<<<dim3(3 * DMODEL / 64, T_TOK / 64), 128, 0, stream>>>(
            z, wt, bqkv + (size_t)d * 3 * DMODEL, nullptr, qkv, nullptr,
            T_TOK, 3 * DMODEL, DMODEL, 0);
        // rope + layouts
        rope_kernel<<<dim3(T_TOK, NHEAD), 96, 0, stream>>>(qkv, cosb, sinb, qp, kp, vt);
        // attention
        attn_kernel<<<dim3(T_TOK / 16, NHEAD), 32, 0, stream>>>(qp, kp, vt, seg, ao);
        // h += ao @ wproj + bproj
        transpose_bf16_kernel<<<dim3(DMODEL / 32, DMODEL / 32), tb, 0, stream>>>(
            wproj + (size_t)d * DMODEL * DMODEL, wt, DMODEL, DMODEL);
        gemm_kernel<<<dim3(DMODEL / 64, T_TOK / 64), 128, 0, stream>>>(
            ao, wt, bproj + (size_t)d * DMODEL, h, h, nullptr,
            T_TOK, DMODEL, DMODEL, 0);
        // ln2 -> z
        ln_kernel<<<T_TOK, 256, 0, stream>>>(h, ln2w + d * DMODEL, ln2b + d * DMODEL, z, DMODEL);
        // g = gelu_tanh(z @ wfc1 + bfc1)
        transpose_bf16_kernel<<<dim3(DMODEL / 32, FFDIM / 32), tb, 0, stream>>>(
            wfc1 + (size_t)d * DMODEL * FFDIM, wt, DMODEL, FFDIM);
        gemm_kernel<<<dim3(FFDIM / 64, T_TOK / 64), 128, 0, stream>>>(
            z, wt, bfc1 + (size_t)d * FFDIM, nullptr, nullptr, g,
            T_TOK, FFDIM, DMODEL, 1);
        // h += g @ wfc2 + bfc2
        transpose_bf16_kernel<<<dim3(FFDIM / 32, DMODEL / 32), tb, 0, stream>>>(
            wfc2 + (size_t)d * FFDIM * DMODEL, wt, FFDIM, DMODEL);
        gemm_kernel<<<dim3(DMODEL / 64, T_TOK / 64), 128, 0, stream>>>(
            g, wt, bfc2 + (size_t)d * DMODEL, h, h, nullptr,
            T_TOK, DMODEL, FFDIM, 0);
    }

    // merger: ln over D rows; z[T][D] bf16 is contiguously [512][4608]
    ln_kernel<<<T_TOK, 256, 0, stream>>>(h, mlnw, mlnb, z, DMODEL);
    transpose_bf16_kernel<<<dim3(MERGED_D / 32, MERGED_D / 32), tb, 0, stream>>>(
        mw1, wt, MERGED_D, MERGED_D);
    gemm_kernel<<<dim3(MERGED_D / 64, 512 / 64), 128, 0, stream>>>(
        z, wt, mb1, nullptr, nullptr, g, 512, MERGED_D, MERGED_D, 2);
    transpose_bf16_kernel<<<dim3(MERGED_D / 32, OUT_D / 32), tb, 0, stream>>>(
        mw2, wt, MERGED_D, OUT_D);
    gemm_kernel<<<dim3(OUT_D / 64, 512 / 64), 128, 0, stream>>>(
        g, wt, mb2, nullptr, (float*)d_out, nullptr, 512, OUT_D, MERGED_D, 0);
}
